// MoEBlock_52097953300681
// MI455X (gfx1250) — matrix operand order repaired and checked by reading, not compiled, not run on hardware
//
#include <hip/hip_runtime.h>
#include <math.h>

// MI455X / gfx1250, wave32. MoE block: router + top-2 gather + grouped bf16 WMMA GEMMs.
// d_out = [T*D f32 out][T*E f32 router_logits].
// Workspace: counts(E i32) | lists(E*T i32, value = token*2+k) | wlist(E*T f32)
//            | h bf16 [T*K rows x H] | ysc f32 [T*K x D]   (~385 MB)

typedef __attribute__((ext_vector_type(16))) __bf16 v16bf;
typedef __attribute__((ext_vector_type(8)))  float  v8f;

#define TOPK 2
#define BM 128
#define BN 128
#define BK 32
#define LDA 48   // A tile LDS row stride (elements), 96B rows keep 16B frag alignment
#define LDB 136  // B tile LDS row stride (elements), 272B rows keep 16B frag alignment

__device__ __forceinline__ float gelu_exact(float x) {
    return 0.5f * x * (1.0f + erff(x * 0.70710678118654752440f));
}

// ---------------------------------------------------------------- init
__global__ void init_counts_kernel(int* counts, int E) {
    if (threadIdx.x < (unsigned)E) counts[threadIdx.x] = 0;
}

// ---------------------------------------------------------------- router
// One wave (32 lanes) per token: logits = x[t] @ rw, softmax, top-2,
// append (token,k) to per-expert lists. E == 8 for this problem.
__global__ __launch_bounds__(256)
void router_kernel(const float* __restrict__ x, const float* __restrict__ rw,
                   float* __restrict__ logits_out,
                   int* __restrict__ counts, int* __restrict__ lists,
                   float* __restrict__ wlist, int T, int D) {
    const int wave = threadIdx.x >> 5;
    const int lane = threadIdx.x & 31;
    const int t = blockIdx.x * (blockDim.x >> 5) + wave;
    if (t >= T) return;

    float acc[8] = {0.f,0.f,0.f,0.f,0.f,0.f,0.f,0.f};
    const float* xr = x + (size_t)t * D;
    for (int d = lane; d < D; d += 32) {
        float xv = xr[d];
        const float4* r4 = (const float4*)(rw + (size_t)d * 8);
        float4 a = r4[0], b = r4[1];
        acc[0] += xv * a.x; acc[1] += xv * a.y; acc[2] += xv * a.z; acc[3] += xv * a.w;
        acc[4] += xv * b.x; acc[5] += xv * b.y; acc[6] += xv * b.z; acc[7] += xv * b.w;
    }
#pragma unroll
    for (int off = 16; off > 0; off >>= 1)
#pragma unroll
        for (int e = 0; e < 8; ++e)
            acc[e] += __shfl_xor(acc[e], off, 32);

    if (lane == 0) {
        float* lo = logits_out + (size_t)t * 8;
#pragma unroll
        for (int e = 0; e < 8; ++e) lo[e] = acc[e];

        float m = acc[0];
#pragma unroll
        for (int e = 1; e < 8; ++e) m = fmaxf(m, acc[e]);
        float p[8]; float s = 0.f;
#pragma unroll
        for (int e = 0; e < 8; ++e) { p[e] = expf(acc[e] - m); s += p[e]; }
        float inv = 1.0f / s;

        int e1 = 0; float v1 = p[0];
#pragma unroll
        for (int e = 1; e < 8; ++e) if (p[e] > v1) { v1 = p[e]; e1 = e; }
        int e2 = -1; float v2 = -1.f;
#pragma unroll
        for (int e = 0; e < 8; ++e) if (e != e1 && p[e] > v2) { v2 = p[e]; e2 = e; }
        v1 *= inv; v2 *= inv;   // softmax probs of the top-2 (not renormalized), as in reference

        int p1 = atomicAdd(&counts[e1], 1);
        lists[e1 * T + p1] = t * 2 + 0; wlist[e1 * T + p1] = v1;
        int p2 = atomicAdd(&counts[e2], 1);
        lists[e2 * T + p2] = t * 2 + 1; wlist[e2 * T + p2] = v2;
    }
}

// ---------------------------------------------------------------- GEMM1: h = gelu(Xg @ W1[e] + b1[e])
__global__ __launch_bounds__(256)
void gemm1_kernel(const float* __restrict__ x, const float* __restrict__ w1,
                  const float* __restrict__ b1,
                  const int* __restrict__ counts, const int* __restrict__ lists,
                  unsigned short* __restrict__ h_u16, int T, int D, int H) {
    __bf16* __restrict__ h = (__bf16*)h_u16;
    const int e = blockIdx.z;

    __shared__ int cnt_s, hbase_s;
    __shared__ __bf16 As[BM * LDA];
    __shared__ __bf16 Bs[BK * LDB];
    __shared__ int toks_s[BM];

    if (threadIdx.x == 0) {
        int base = 0;
        for (int i = 0; i < e; ++i) base += counts[i];
        cnt_s = counts[e]; hbase_s = base;
    }
    __syncthreads();
    const int cnt = cnt_s;
    const int m0 = blockIdx.y * BM;
    if (m0 >= cnt) return;
    const int hbase = hbase_s;
    const int n0 = blockIdx.x * BN;

    if (threadIdx.x < BM) {
        int g = m0 + (int)threadIdx.x;
        toks_s[threadIdx.x] = (g < cnt) ? (lists[e * T + g] >> 1) : -1;
    }
    __syncthreads();

    const int tid  = threadIdx.x;
    const int lane = tid & 31;
    const int wv   = tid >> 5;
    const int wm   = (wv >> 1) * 32;  // 4 waves over M
    const int wn   = (wv & 1) * 64;   // 2 waves over N

    v8f acc[2][4] = {};

    const int rowA = tid >> 1, colA = (tid & 1) * 16;   // 2 threads per A row
    const int rowB = tid >> 3, colB = (tid & 7) * 16;   // 8 threads per B row
    const size_t w1e = (size_t)e * D * H;

    for (int k0 = 0; k0 < D; k0 += BK) {
        // stage A: gathered token rows, f32 -> bf16
        {
            int tok = toks_s[rowA];
            __bf16* dst = &As[rowA * LDA + colA];
            if (tok >= 0) {
                const float* src = x + (size_t)tok * D + k0 + colA;
#pragma unroll
                for (int j = 0; j < 16; ++j) dst[j] = (__bf16)src[j];
            } else {
#pragma unroll
                for (int j = 0; j < 16; ++j) dst[j] = (__bf16)0.0f;
            }
        }
        // stage B: W1[e][k0+rowB][n0+colB..], f32 -> bf16
        {
            const float* src = w1 + w1e + (size_t)(k0 + rowB) * H + n0 + colB;
            __bf16* dst = &Bs[rowB * LDB + colB];
#pragma unroll
            for (int j = 0; j < 16; ++j) dst[j] = (__bf16)src[j];
            if (k0 + BK < D) __builtin_prefetch(src + (size_t)BK * H, 0, 0);
        }
        __syncthreads();

        // A frag: lanes 0-15 rows M, K halves {0..7,16..23}; lanes 16-31 K halves {8..15,24..31}
        const int arow = lane & 15;
        const int ak0 = (lane < 16) ? 0 : 8;
        const int ak1 = ak0 + 16;
        v16bf af[2];
#pragma unroll
        for (int mi = 0; mi < 2; ++mi) {
            const __bf16* ap = &As[(wm + mi * 16 + arow) * LDA];
            ((uint4*)&af[mi])[0] = *(const uint4*)(ap + ak0);
            ((uint4*)&af[mi])[1] = *(const uint4*)(ap + ak1);
        }
#pragma unroll
        for (int ni = 0; ni < 4; ++ni) {
            // B frag: lane L holds N=L%16, K in same table order as A
            v16bf bfrag;
            const __bf16* bp = &Bs[wn + ni * 16 + (lane & 15)];
#pragma unroll
            for (int s = 0; s < 16; ++s) bfrag[s] = bp[(s < 8 ? ak0 + s : ak1 + s - 8) * LDB];
#pragma unroll
            for (int mi = 0; mi < 2; ++mi)
                acc[mi][ni] = __builtin_amdgcn_wmma_f32_16x16x32_bf16(
                    false, af[mi], false, bfrag, (short)0, acc[mi][ni], false, false);
        }
        __syncthreads();
    }

    // epilogue: +b1, exact GELU, store bf16 h (compacted rows per expert)
    const int crow = (lane < 16) ? 0 : 8;  // C: VGPR r -> M=r (lanes 0-15) / M=r+8 (lanes 16-31)
    const int ccol = lane & 15;
#pragma unroll
    for (int mi = 0; mi < 2; ++mi)
#pragma unroll
        for (int ni = 0; ni < 4; ++ni) {
            const int cg = n0 + wn + ni * 16 + ccol;
            const float bb = b1[(size_t)e * H + cg];
#pragma unroll
            for (int r = 0; r < 8; ++r) {
                int g = m0 + wm + mi * 16 + crow + r;
                if (g < cnt)
                    h[(size_t)(hbase + g) * H + cg] = (__bf16)gelu_exact(acc[mi][ni][r] + bb);
            }
        }
}

// ---------------------------------------------------------------- GEMM2: ysc[t,k] = w * (h @ W2[e] + b2[e])
__global__ __launch_bounds__(256)
void gemm2_kernel(const unsigned short* __restrict__ h_u16, const float* __restrict__ w2,
                  const float* __restrict__ b2,
                  const int* __restrict__ counts, const int* __restrict__ lists,
                  const float* __restrict__ wlist,
                  float* __restrict__ ysc, int T, int D, int H) {
    const __bf16* __restrict__ h = (const __bf16*)h_u16;
    const int e = blockIdx.z;

    __shared__ int cnt_s, hbase_s;
    __shared__ __bf16 As[BM * LDA];
    __shared__ __bf16 Bs[BK * LDB];
    __shared__ int   toks_s[BM];
    __shared__ float wts_s[BM];

    if (threadIdx.x == 0) {
        int base = 0;
        for (int i = 0; i < e; ++i) base += counts[i];
        cnt_s = counts[e]; hbase_s = base;
    }
    __syncthreads();
    const int cnt = cnt_s;
    const int m0 = blockIdx.y * BM;
    if (m0 >= cnt) return;
    const int hbase = hbase_s;
    const int n0 = blockIdx.x * BN;

    if (threadIdx.x < BM) {
        int g = m0 + (int)threadIdx.x;
        toks_s[threadIdx.x] = (g < cnt) ? lists[e * T + g] : 0;
        wts_s[threadIdx.x]  = (g < cnt) ? wlist[e * T + g] : 0.f;
    }
    __syncthreads();

    const int tid  = threadIdx.x;
    const int lane = tid & 31;
    const int wv   = tid >> 5;
    const int wm   = (wv >> 1) * 32;
    const int wn   = (wv & 1) * 64;

    v8f acc[2][4] = {};

    const int rowA = tid >> 1, colA = (tid & 1) * 16;
    const int rowB = tid >> 3, colB = (tid & 7) * 16;
    const size_t w2e = (size_t)e * H * D;

    for (int k0 = 0; k0 < H; k0 += BK) {
        // stage A: h rows already bf16 and contiguous -> straight 16B copies
        {
            int g = m0 + rowA;
            uint4 z = {0u, 0u, 0u, 0u};
            uint4 d0 = z, d1 = z;
            if (g < cnt) {
                const uint4* src = (const uint4*)(h + (size_t)(hbase + g) * H + k0 + colA);
                d0 = src[0]; d1 = src[1];
            }
            *(uint4*)&As[rowA * LDA + colA]     = d0;
            *(uint4*)&As[rowA * LDA + colA + 8] = d1;
        }
        // stage B: W2[e][k][n] f32 -> bf16
        {
            const float* src = w2 + w2e + (size_t)(k0 + rowB) * D + n0 + colB;
            __bf16* dst = &Bs[rowB * LDB + colB];
#pragma unroll
            for (int j = 0; j < 16; ++j) dst[j] = (__bf16)src[j];
            if (k0 + BK < H) __builtin_prefetch(src + (size_t)BK * D, 0, 0);
        }
        __syncthreads();

        const int arow = lane & 15;
        const int ak0 = (lane < 16) ? 0 : 8;
        const int ak1 = ak0 + 16;
        v16bf af[2];
#pragma unroll
        for (int mi = 0; mi < 2; ++mi) {
            const __bf16* ap = &As[(wm + mi * 16 + arow) * LDA];
            ((uint4*)&af[mi])[0] = *(const uint4*)(ap + ak0);
            ((uint4*)&af[mi])[1] = *(const uint4*)(ap + ak1);
        }
#pragma unroll
        for (int ni = 0; ni < 4; ++ni) {
            v16bf bfrag;
            const __bf16* bp = &Bs[wn + ni * 16 + (lane & 15)];
#pragma unroll
            for (int s = 0; s < 16; ++s) bfrag[s] = bp[(s < 8 ? ak0 + s : ak1 + s - 8) * LDB];
#pragma unroll
            for (int mi = 0; mi < 2; ++mi)
                acc[mi][ni] = __builtin_amdgcn_wmma_f32_16x16x32_bf16(
                    false, af[mi], false, bfrag, (short)0, acc[mi][ni], false, false);
        }
        __syncthreads();
    }

    // epilogue: +b2, scale by combine weight, deterministic scatter into (token,k) slot
    const int crow = (lane < 16) ? 0 : 8;
    const int ccol = lane & 15;
#pragma unroll
    for (int mi = 0; mi < 2; ++mi)
#pragma unroll
        for (int ni = 0; ni < 4; ++ni) {
            const int cg = n0 + wn + ni * 16 + ccol;
            const float bb = b2[(size_t)e * D + cg];
#pragma unroll
            for (int r = 0; r < 8; ++r) {
                int rl = wm + mi * 16 + crow + r;
                int g = m0 + rl;
                if (g < cnt) {
                    int tk = toks_s[rl];
                    ysc[(size_t)tk * D + cg] = wts_s[rl] * (acc[mi][ni][r] + bb);
                }
            }
        }
}

// ---------------------------------------------------------------- combine: out = bias + slot0 + slot1
__global__ void combine_kernel(const float* __restrict__ ysc, const float* __restrict__ bias,
                               float* __restrict__ out, int T, int D) {
    size_t i = (size_t)blockIdx.x * blockDim.x + threadIdx.x;
    if (i >= (size_t)T * D) return;
    int d = (int)(i % D);
    size_t t = i / D;
    out[i] = bias[d] + ysc[t * 2 * (size_t)D + d] + ysc[(t * 2 + 1) * (size_t)D + d];
}

// ---------------------------------------------------------------- launch
extern "C" void kernel_launch(void* const* d_in, const int* in_sizes, int n_in,
                              void* d_out, int out_size, void* d_ws, size_t ws_size,
                              hipStream_t stream) {
    const float* x    = (const float*)d_in[0];
    const float* rw   = (const float*)d_in[1];
    const float* w1   = (const float*)d_in[2];
    const float* b1   = (const float*)d_in[3];
    const float* w2   = (const float*)d_in[4];
    const float* b2   = (const float*)d_in[5];
    const float* bias = (const float*)d_in[6];

    const int D = in_sizes[6];             // 1024
    const int E = in_sizes[1] / D;         // 8
    const int T = in_sizes[0] / D;         // 16384
    const int H = in_sizes[3] / E;         // 4096

    float* out = (float*)d_out;
    float* logits_out = out + (size_t)T * D;

    char* ws = (char*)d_ws;
    int*   counts = (int*)ws;                                   // E i32
    int*   lists  = (int*)(ws + 256);                           // E*T i32
    float* wlist  = (float*)(ws + 256 + (size_t)E * T * 4);     // E*T f32
    size_t off = 256 + (size_t)E * T * 8;
    off = (off + 255) & ~(size_t)255;
    unsigned short* h = (unsigned short*)(ws + off);            // (T*K)*H bf16
    off += (size_t)T * TOPK * H * 2;
    off = (off + 255) & ~(size_t)255;
    float* ysc = (float*)(ws + off);                            // (T*K)*D f32

    init_counts_kernel<<<1, 32, 0, stream>>>(counts, E);
    router_kernel<<<(T + 7) / 8, 256, 0, stream>>>(x, rw, logits_out, counts, lists, wlist, T, D);

    dim3 g1(H / BN, (T + BM - 1) / BM, E);
    gemm1_kernel<<<g1, 256, 0, stream>>>(x, w1, b1, counts, lists, h, T, D, H);

    dim3 g2(D / BN, (T + BM - 1) / BM, E);
    gemm2_kernel<<<g2, 256, 0, stream>>>(h, w2, b2, counts, lists, wlist, ysc, T, D, H);

    size_t tot = (size_t)T * D;
    combine_kernel<<<(unsigned)((tot + 255) / 256), 256, 0, stream>>>(ysc, bias, out, T, D);
}
